// BidderStrategy_7773890805977
// MI455X (gfx1250) — matrix operand order repaired and checked by reading, not compiled, not run on hardware
//
#include <hip/hip_runtime.h>
#include <cstdint>

typedef __attribute__((ext_vector_type(16))) _Float16 v16h;
typedef __attribute__((ext_vector_type(8)))  float    v8f;
typedef __attribute__((ext_vector_type(2)))  _Float16 v2h;

#define NACT  12
#define HDIM  200
#define HPAD  224      // 14 tiles of 16 rows -> 7 tile-PAIRS per n (full K=32 in WMMA2)
#define NTILE 14
#define NPAIR 7
#define W2PN  112      // dwords per n in the paired w2 table: 7 pairs * 16 dwords
#define BTOT  131072
#define ZLEN  216      // zero-feed dwords; covers max DS imm offsets used below

union VecU {
  uint32_t u[8];
  v16h     h;
};

static __device__ __forceinline__ uint32_t pk_f16(float a, float b) {
  v2h p;
  p[0] = (_Float16)a;
  p[1] = (_Float16)b;
  return __builtin_bit_cast(uint32_t, p);   // fuses to v_cvt_pk_f16_f32
}

// cvt is monotonic with cvt(0)=0, so relu commutes with f16 rounding.
static __device__ __forceinline__ uint32_t relu_pk(float a, float b) {
  uint32_t u = pk_f16(a, b);
  uint32_t r;
  asm("v_pk_max_num_f16 %0, %1, 0" : "=v"(r) : "v"(u));
  return r;
}

static __device__ __forceinline__ v8f wmma1(const VecU& A, const VecU& B) {
  const v8f cz = {};
  return __builtin_amdgcn_wmma_f32_16x16x32_f16(false, A.h, false, B.h,
                                                (short)0, cz, false, false);
}

__global__ __launch_bounds__(256) void bidder_mlp_wmma(
    const float* __restrict__ inp,
    const float* __restrict__ w1,
    const float* __restrict__ b1,
    const float* __restrict__ w2,
    const float* __restrict__ b2,
    const float* __restrict__ w3,
    float* __restrict__ out)
{
  __shared__ __align__(16) uint32_t s_w1b1[NACT * HPAD];  // pk(w1[h], b1[h]), zero-padded
  __shared__ __align__(16) uint32_t s_w2p [NACT * W2PN];  // w2 in tile-pair blocks (see below)
  __shared__ __align__(16) uint32_t s_zero[ZLEN];         // zero feed for inactive lanes

  const int tid = threadIdx.x;

  for (int i = tid; i < NACT * HPAD; i += 256) {
    const int n = i / HPAD, h = i - n * HPAD;
    s_w1b1[i] = (h < HDIM) ? pk_f16(w1[n * HDIM + h], b1[n * HDIM + h]) : 0u;
  }
  // Per tile-pair block of 16 dwords:
  //  dwords  0- 3: pk pairs of rows tp*32 + 0..7    (lane0  VGPR0-3, K=0..7)
  //  dwords  4- 7: pk pairs of rows tp*32 + 16..23  (lane0  VGPR4-7, K=16..23)
  //  dwords  8-11: pk pairs of rows tp*32 + 8..15   (lane16 VGPR0-3, K=8..15)
  //  dwords 12-15: pk pairs of rows tp*32 + 24..31  (lane16 VGPR4-7, K=24..31)
  for (int i = tid; i < NACT * W2PN; i += 256) {
    const int n  = i / W2PN, p = i - n * W2PN;
    const int tp = p >> 4, q = p & 15, g = q >> 2, ii = q & 3;
    const int row = tp * 32 + (g & 1) * 16 + (g >> 1) * 8 + 2 * ii;
    const float a = (row     < HDIM) ? w2[n * HDIM + row    ] : 0.0f;
    const float c = (row + 1 < HDIM) ? w2[n * HDIM + row + 1] : 0.0f;
    s_w2p[i] = pk_f16(a, c);
  }
  if (tid < ZLEN) s_zero[tid] = 0u;
  if (blockIdx.x == 0 && tid < NACT) out[tid] = w3[tid];   // alpha = w3[:,0]
  __syncthreads();

  const int lane = tid & 31;
  const int wave = tid >> 5;
  const int b0   = (blockIdx.x * 8 + wave) * 16;   // 16 batch elems per wave

  // B1 (32x16 f16), loop-invariant: col b = [x_b, 1, 0, ...]
  float x = 0.0f;
  if (lane < 16) x = inp[b0 + lane];
  VecU B1;
  #pragma unroll
  for (int i = 0; i < 8; ++i) B1.u[i] = 0u;
  B1.u[0] = (lane < 16) ? pk_f16(x, 1.0f) : 0u;

  const v8f czero = {};

  for (int n = 0; n < NACT; ++n) {
    // Base pointers chosen once per n; pair index folds into DS imm offsets.
    const uint32_t* a1base = (lane < 16) ? (s_w1b1 + n * HPAD + lane) : s_zero;
    const uint4*    a2base =
        (lane == 0)  ? (const uint4*)(s_w2p + n * W2PN)
      : (lane == 16) ? (const uint4*)(s_w2p + n * W2PN + 8)
                     : (const uint4*)s_zero;

    v8f acc0 = czero;
    v8f acc1 = czero;

    VecU A1;
    #pragma unroll
    for (int i = 0; i < 8; ++i) A1.u[i] = 0u;

    // Pipeline prologue: layer-1 WMMAs for tile pair 0.
    A1.u[0] = a1base[0];
    v8f d1a = wmma1(A1, B1);
    A1.u[0] = a1base[16];
    v8f d1b = wmma1(A1, B1);

    #pragma unroll
    for (int tp = 0; tp < NPAIR; ++tp) {
      v8f d1an = czero, d1bn = czero;
      if (tp + 1 < NPAIR) {
        A1.u[0] = a1base[(2 * tp + 2) * 16];
        d1an = wmma1(A1, B1);
        A1.u[0] = a1base[(2 * tp + 3) * 16];
        d1bn = wmma1(A1, B1);
      }

      // A2: lane0 = w2 rows 0-7 of tiles a and b, lane16 = rows 8-15 of each, others zero.
      const uint4 q0 = a2base[tp * 4];
      const uint4 q1 = a2base[tp * 4 + 1];
      VecU A2;
      A2.u[0] = q0.x; A2.u[1] = q0.y; A2.u[2] = q0.z; A2.u[3] = q0.w;
      A2.u[4] = q1.x; A2.u[5] = q1.y; A2.u[6] = q1.z; A2.u[7] = q1.w;

      // B2: lower oct = relu(tile a), upper oct = relu(tile b). Full K=32 used.
      VecU B2;
      B2.u[0] = relu_pk(d1a[0], d1a[1]);
      B2.u[1] = relu_pk(d1a[2], d1a[3]);
      B2.u[2] = relu_pk(d1a[4], d1a[5]);
      B2.u[3] = relu_pk(d1a[6], d1a[7]);
      B2.u[4] = relu_pk(d1b[0], d1b[1]);
      B2.u[5] = relu_pk(d1b[2], d1b[3]);
      B2.u[6] = relu_pk(d1b[4], d1b[5]);
      B2.u[7] = relu_pk(d1b[6], d1b[7]);

      if (tp & 1) {
        acc1 = __builtin_amdgcn_wmma_f32_16x16x32_f16(
            false, A2.h, false, B2.h, (short)0, acc1, false, false);
      } else {
        acc0 = __builtin_amdgcn_wmma_f32_16x16x32_f16(
            false, A2.h, false, B2.h, (short)0, acc0, false, false);
      }
      d1a = d1an;
      d1b = d1bn;
    }

    if (lane < 16) {
      out[12 + (size_t)n * BTOT + b0 + lane] = acc0[0] + acc1[0] + b2[n];
    }
  }
}

extern "C" void kernel_launch(void* const* d_in, const int* in_sizes, int n_in,
                              void* d_out, int out_size, void* d_ws, size_t ws_size,
                              hipStream_t stream) {
  (void)in_sizes; (void)n_in; (void)out_size; (void)d_ws; (void)ws_size;
  const float* inp = (const float*)d_in[0];
  const float* w1  = (const float*)d_in[1];
  const float* b1  = (const float*)d_in[2];
  const float* w2  = (const float*)d_in[3];
  const float* b2  = (const float*)d_in[4];
  const float* w3  = (const float*)d_in[5];
  float* out = (float*)d_out;

  const int blocks = BTOT / 128;   // 8 waves/block * 16 b's per wave
  bidder_mlp_wmma<<<blocks, 256, 0, stream>>>(inp, w1, b1, w2, b2, w3, out);
}
